// Multi_Head_Attention_36962488549510
// MI455X (gfx1250) — matrix operand order repaired and checked by reading, not compiled, not run on hardware
//
#include <hip/hip_runtime.h>

// MI455X / gfx1250, wave32. All matmuls on v_wmma_f32_16x16x32_f16.
typedef _Float16 v16h __attribute__((ext_vector_type(16)));
typedef float    v8f  __attribute__((ext_vector_type(8)));

namespace {
constexpr int Bn = 8;      // batch
constexpr int C  = 256;    // channels
constexpr int N  = 1024;   // H*W
constexpr int NH = 8;      // heads
constexpr int Dh = 32;     // head dim
constexpr float SCALE = 1.0f / 32.0f;   // 1/sqrt(N), per reference quirk

// A-fragment (16x32 f16, MxK) K index for element e of the v16h (contiguous order):
// lanes 0-15 hold K 0..15, lanes 16-31 hold K 16..31 (matches B-fragment).
__device__ __forceinline__ int ak(int e, int half) {
    return e + 16 * half;
}
// B-fragment (32x16 f16, KxN) used with ak, contiguous: lanes 0-15 hold K 0..15, lanes 16-31 hold K 16..31.
} // namespace

// ---------------------------------------------------------------------------
// Kernel 1: q = x^T @ Wq^T + bq ; kv = y^T @ Wv^T + bv.  One wave per
// (sel, b, 16-row n-tile, 16-col co-tile). Output stored per-head as f16:
// qh/kvh[b, h, n, dd] with h = co>>5, dd = co&31.
// ---------------------------------------------------------------------------
__global__ void k_proj(const float* __restrict__ x, const float* __restrict__ y,
                       const float* __restrict__ Wq, const float* __restrict__ bq,
                       const float* __restrict__ Wv, const float* __restrict__ bv,
                       _Float16* __restrict__ qh, _Float16* __restrict__ kvh) {
    int gw = blockIdx.x * (blockDim.x >> 5) + (threadIdx.x >> 5);
    int ct = gw & 15; gw >>= 4;
    int nt = gw & 63; gw >>= 6;
    int b  = gw & 7;  gw >>= 3;
    int sel = gw;                                   // 0: q from x, 1: kv from y

    const float* src  = sel ? y   : x;
    const float* W    = sel ? Wv  : Wq;
    const float* bias = sel ? bv  : bq;
    _Float16*    dst  = sel ? kvh : qh;

    const int lane = threadIdx.x & 31;
    const int half = lane >> 4;
    const int l16  = lane & 15;
    const int n0 = nt * 16, co0 = ct * 16;

    v8f acc = {};
    for (int k0 = 0; k0 < C; k0 += 32) {
        // speculative prefetch of next K-chunk (global_prefetch_b8)
        if (k0 + 32 < C) {
            __builtin_prefetch(&src[((size_t)b * C + (k0 + 32 + lane)) * N + n0], 0, 1);
            __builtin_prefetch(&W[(size_t)(co0 + l16) * C + k0 + 32 + 16 * half], 0, 1);
        }
        v16h a, bm;
#pragma unroll
        for (int e = 0; e < 16; ++e) {
            // A[m=n-local, k=c]: xf[b,n,c] = x[b, c, n] (x is [B,C,H,W])
            a[e]  = (_Float16)src[((size_t)b * C + (k0 + ak(e, half))) * N + n0 + l16];
            // B[k=c, n=co] = W[co, c]  (contiguous in e -> vector loads)
            bm[e] = (_Float16)W[(size_t)(co0 + l16) * C + k0 + e + 16 * half];
        }
        acc = __builtin_amdgcn_wmma_f32_16x16x32_f16(false, a, false, bm, (short)0, acc,
                                                     false, false);
    }
    const float bb = bias[co0 + l16];
    const int h  = co0 >> 5;               // 16 | co0 => head constant per tile
    const int dd = (co0 & 31) + l16;
#pragma unroll
    for (int r = 0; r < 8; ++r) {          // D: M = r + 8*half, N = l16
        int m = r + 8 * half;
        dst[(((size_t)b * NH + h) * N + n0 + m) * Dh + dd] = (_Float16)(acc[r] + bb);
    }
}

// ---------------------------------------------------------------------------
// Kernel 2: flash attention per (b, h, 16-row j-tile).
//   S = Q(16i x 32d) @ Kv_j^T(32d x 16j) -> softmax over i (M axis) per j,
//   O(16j x 32d) += P^T(16j x 32i) @ Kv_i(32i x 32d).
// The exp'd S tiles in D layout feed the P^T A-fragment with zero shuffles.
// ---------------------------------------------------------------------------
__global__ void k_attn(const _Float16* __restrict__ qh, const _Float16* __restrict__ kvh,
                       float* __restrict__ obuf) {
    int gw = blockIdx.x * (blockDim.x >> 5) + (threadIdx.x >> 5);
    int jt = gw & 63; gw >>= 6;
    int h  = gw & 7;  gw >>= 3;
    int b  = gw;

    const int lane = threadIdx.x & 31;
    const int half = lane >> 4;
    const int l16  = lane & 15;
    const int j0 = jt * 16;

    const _Float16* Q  = qh  + ((size_t)b * NH + h) * (size_t)N * Dh;
    const _Float16* KV = kvh + ((size_t)b * NH + h) * (size_t)N * Dh;

    // Fixed B-fragment: B[k=d, n=j] = Kv[j0+n, d]
    v16h bj;
#pragma unroll
    for (int e = 0; e < 16; ++e)
        bj[e] = KV[(size_t)(j0 + l16) * Dh + e + 16 * half];

    float mrow = -__builtin_inff();   // running max  (for j = lane&15, both halves)
    float lrow = 0.0f;                // running sum
    v8f o0 = {}, o1 = {};             // O tile: N=dd (0..15 / 16..31), M=j=r+8*half

    for (int i0 = 0; i0 < N; i0 += 32) {
        // prefetch next i-chunk of Q and KV while this chunk computes
        if (i0 + 32 < N) {
            __builtin_prefetch(&Q[(size_t)(i0 + 32 + lane) * Dh], 0, 1);
            __builtin_prefetch(&KV[(size_t)(i0 + 32 + lane) * Dh], 0, 1);
        }
        // Two S tiles (i0..i0+15, i0+16..i0+31): A[m=i-local, k=d] = Q[i, d]
        v16h a1, a2;
#pragma unroll
        for (int e = 0; e < 16; ++e) {
            int k = ak(e, half);
            a1[e] = Q[(size_t)(i0 + l16) * Dh + k];
            a2[e] = Q[(size_t)(i0 + 16 + l16) * Dh + k];
        }
        v8f z = {};
        v8f s1 = __builtin_amdgcn_wmma_f32_16x16x32_f16(false, a1, false, bj, (short)0, z, false, false);
        v8f s2 = __builtin_amdgcn_wmma_f32_16x16x32_f16(false, a2, false, bj, (short)0, z, false, false);

        // chunk max over i (in-lane 16 regs, then cross-half combine)
        float cm = -__builtin_inff();
#pragma unroll
        for (int r = 0; r < 8; ++r) {
            s1[r] *= SCALE; s2[r] *= SCALE;
            cm = fmaxf(cm, fmaxf(s1[r], s2[r]));
        }
        cm = fmaxf(cm, __shfl_xor(cm, 16, 32));
        float mnew  = fmaxf(mrow, cm);
        float alpha = __expf(mrow - mnew);       // 0 on first chunk (exp(-inf))
        lrow *= alpha;
#pragma unroll
        for (int r = 0; r < 8; ++r) {            // O reg r is row j = r+8*half
            float al = __shfl(alpha, r + 8 * half, 32);
            o0[r] *= al; o1[r] *= al;
        }
        float ps = 0.0f;
#pragma unroll
        for (int r = 0; r < 8; ++r) {
            s1[r] = __expf(s1[r] - mnew); ps += s1[r];
            s2[r] = __expf(s2[r] - mnew); ps += s2[r];
        }
        ps += __shfl_xor(ps, 16, 32);
        lrow += ps;
        mrow = mnew;

        // P^T A-fragment: lane m=j=l16, K(e)=i-local -> exactly s1[e] / s2[e-8]
        v16h pa;
#pragma unroll
        for (int e = 0; e < 8; ++e) {
            pa[e]     = (_Float16)s1[e];
            pa[e + 8] = (_Float16)s2[e];
        }
        // B[k=i-local, n=dd] = Kv[i0 + k, dd]
        v16h bv0, bv1;
#pragma unroll
        for (int e = 0; e < 16; ++e) {
            size_t row = (size_t)(i0 + ((e < 8) ? e : (e + 8)) + 8 * half) * Dh;
            bv0[e] = KV[row + l16];
            bv1[e] = KV[row + 16 + l16];
        }
        o0 = __builtin_amdgcn_wmma_f32_16x16x32_f16(false, pa, false, bv0, (short)0, o0, false, false);
        o1 = __builtin_amdgcn_wmma_f32_16x16x32_f16(false, pa, false, bv1, (short)0, o1, false, false);
    }

    float* Ob = obuf + ((size_t)b * NH + h) * (size_t)N * Dh;
#pragma unroll
    for (int r = 0; r < 8; ++r) {
        float lj  = __shfl(lrow, r + 8 * half, 32);
        float inv = 1.0f / lj;
        size_t row = (size_t)(j0 + r + 8 * half) * Dh;
        Ob[row + l16]      = o0[r] * inv;
        Ob[row + 16 + l16] = o1[r] * inv;
    }
}

// ---------------------------------------------------------------------------
// Kernel 3: final GEMM with the faithful reinterpret gather:
//   out2[b,n,c] = O[b, n>>7, (n&3)*256 + c, (n>>2)&31];  out = out2 @ Wf^T + bf
// ---------------------------------------------------------------------------
__global__ void k_out(const float* __restrict__ obuf, const float* __restrict__ Wf,
                      const float* __restrict__ bfv, float* __restrict__ out) {
    int gw = blockIdx.x * (blockDim.x >> 5) + (threadIdx.x >> 5);
    int ct = gw & 15; gw >>= 4;
    int nt = gw & 63; gw >>= 6;
    int b  = gw;

    const int lane = threadIdx.x & 31;
    const int half = lane >> 4;
    const int l16  = lane & 15;
    const int n0 = nt * 16, co0 = ct * 16;

    const int nn = n0 + l16;                 // A row m -> n index (lane-fixed)
    const int hn = nn >> 7;
    const int dn = (nn >> 2) & 31;
    const int nbase = (nn & 3) << 8;
    const float* Ob = obuf + ((size_t)b * NH + hn) * (size_t)N * Dh;

    v8f acc = {};
    for (int k0 = 0; k0 < C; k0 += 32) {
        if (k0 + 32 < C) {
            __builtin_prefetch(&Ob[(size_t)(nbase + k0 + 32 + lane) * Dh + dn], 0, 1);
            __builtin_prefetch(&Wf[(size_t)(co0 + l16) * C + k0 + 32 + 16 * half], 0, 1);
        }
        v16h a, bm;
#pragma unroll
        for (int e = 0; e < 16; ++e) {
            a[e]  = (_Float16)Ob[(size_t)(nbase + k0 + ak(e, half)) * Dh + dn];
            bm[e] = (_Float16)Wf[(size_t)(co0 + l16) * C + k0 + e + 16 * half];
        }
        acc = __builtin_amdgcn_wmma_f32_16x16x32_f16(false, a, false, bm, (short)0, acc,
                                                     false, false);
    }
    const float bb = bfv[co0 + l16];
#pragma unroll
    for (int r = 0; r < 8; ++r) {
        int m = r + 8 * half;
        out[((size_t)b * N + n0 + m) * C + co0 + l16] = acc[r] + bb;
    }
}

// ---------------------------------------------------------------------------
// Kernel 4: in-place LayerNorm over C=256 per (b,n) row. One wave per row.
// ---------------------------------------------------------------------------
__global__ void k_ln(float* __restrict__ out, const float* __restrict__ gamma,
                     const float* __restrict__ beta) {
    int row = blockIdx.x * (blockDim.x >> 5) + (threadIdx.x >> 5);
    const int lane = threadIdx.x & 31;
    float* p = out + (size_t)row * C;

    float v[8], s = 0.0f, ss = 0.0f;
#pragma unroll
    for (int k = 0; k < 8; ++k) {
        v[k] = p[k * 32 + lane];
        s  += v[k];
        ss += v[k] * v[k];
    }
#pragma unroll
    for (int m = 16; m >= 1; m >>= 1) {
        s  += __shfl_xor(s,  m, 32);
        ss += __shfl_xor(ss, m, 32);
    }
    float mu   = s  * (1.0f / C);
    float var  = ss * (1.0f / C) - mu * mu;
    float rstd = rsqrtf(var + 1e-5f);
#pragma unroll
    for (int k = 0; k < 8; ++k) {
        int c = k * 32 + lane;
        p[c] = (v[k] - mu) * rstd * gamma[c] + beta[c];
    }
}

// ---------------------------------------------------------------------------
extern "C" void kernel_launch(void* const* d_in, const int* in_sizes, int n_in,
                              void* d_out, int out_size, void* d_ws, size_t ws_size,
                              hipStream_t stream) {
    (void)in_sizes; (void)n_in; (void)out_size; (void)ws_size;
    const float* x  = (const float*)d_in[0];
    const float* y  = (const float*)d_in[1];
    const float* Wq = (const float*)d_in[2];
    const float* bq = (const float*)d_in[3];
    const float* Wv = (const float*)d_in[4];
    const float* bv = (const float*)d_in[5];
    const float* Wf = (const float*)d_in[6];
    const float* bf = (const float*)d_in[7];
    const float* gm = (const float*)d_in[8];
    const float* bt = (const float*)d_in[9];
    float* out = (float*)d_out;

    // workspace: qh (4MB f16) | kvh (4MB f16) | attention out (8MB f32)
    _Float16* qh  = (_Float16*)d_ws;
    _Float16* kvh = qh + (size_t)Bn * NH * N * Dh;
    float*    ob  = (float*)(kvh + (size_t)Bn * NH * N * Dh);

    dim3 blk(256);  // 8 waves per block (wave32)
    k_proj<<<(2 * Bn * 64 * 16) / 8, blk, 0, stream>>>(x, y, Wq, bq, Wv, bv, qh, kvh);
    k_attn<<<(Bn * NH * 64) / 8,     blk, 0, stream>>>(qh, kvh, ob);
    k_out <<<(Bn * 64 * 16) / 8,     blk, 0, stream>>>(ob, Wf, bf, out);
    k_ln  <<<(Bn * N) / 8,           blk, 0, stream>>>(out, gm, bt);
}
